// self_att_33174327394948
// MI455X (gfx1250) — matrix operand order repaired and checked by reading, not compiled, not run on hardware
//
#include <hip/hip_runtime.h>

// MI455X / gfx1250, wave32. WMMA f32_16x16x32_f16 everywhere a contraction is big.
// Pipeline (all on `stream`):
//   zero(G) -> dwconv(channel-major y_c) -> dwconv(pixel-major y_t)
//   -> Gram G = Y*Y^T (WMMA, split-K, f32 atomics)
//   -> U=Wq*G, Vh=Wk*G -> per-row scale+softmax S -> T=P*S -> M=T*Wv (f16)
//   -> out = M*Y (WMMA, M-tile staged to LDS via async-to-LDS DMA)

#define B_   8
#define C_   64
#define H_   256
#define W_   256
#define HW_  65536
#define EPSN 1e-12f

typedef __attribute__((ext_vector_type(16))) _Float16 v16h;
typedef __attribute__((ext_vector_type(8)))  _Float16 v8h;
typedef __attribute__((ext_vector_type(8)))  float    v8f;

// ---- WMMA fragment helpers (contiguous k = 16*(lane/16)+e in A and B, wave32) -------------------

// A-matrix 16x32 f16, row-major source with leading dim `ld` (halves).
// lane<16 holds K=0..15 of row (lane&15); lane>=16 holds K=16..31.
__device__ __forceinline__ v16h load_a_frag(const _Float16* A, long ld,
                                            int row0, int k0, int lane) {
    int m  = lane & 15;
    int kb = (lane < 16) ? 0 : 16;
    const _Float16* p = A + (long)(row0 + m) * ld + (k0 + kb);
    union { v16h v; v8h h[2]; } u;
    u.h[0] = *(const v8h*)(p);        // K = kb+0..7
    u.h[1] = *(const v8h*)(p + 8);    // K = kb+8..15
    return u.v;
}

// B-matrix 32x16 f16 (KxN), loaded from an N-major (NxK row-major) source:
// lane holds column n=(lane&15), K=(lane<16?0:16)+e -> 16 contiguous halves.
__device__ __forceinline__ v16h load_b_frag(const _Float16* Bm, long ld,
                                            int n0, int k0, int lane) {
    int n  = lane & 15;
    int kb = (lane < 16) ? 0 : 16;
    return *(const v16h*)(Bm + (long)(n0 + n) * ld + (k0 + kb));
}

__device__ __forceinline__ v8f wmma_f16(v16h a, v16h b, v8f c) {
    return __builtin_amdgcn_wmma_f32_16x16x32_f16(false, a, false, b,
                                                  (short)0, c, false, false);
}

// ---------------------------------------------------------------------------

__global__ void zero_f32_kernel(float* p, int n) {
    int i = blockIdx.x * blockDim.x + threadIdx.x;
    if (i < n) p[i] = 0.0f;
}

// Depthwise 3x3, channel-major output y_c[(b,c,hw)] f16. One thread per (w).
__global__ __launch_bounds__(256) void dwc_kernel(const float* __restrict__ x,
                                                  const float* __restrict__ dw,
                                                  _Float16* __restrict__ y_c) {
    int w  = threadIdx.x;
    int h  = blockIdx.x;
    int c  = blockIdx.y;
    int bi = blockIdx.z;
    const float* xb = x + ((long)bi * C_ + c) * HW_;
    float acc = 0.0f;
#pragma unroll
    for (int dy = 0; dy < 3; ++dy) {
        int yy = h + dy - 1;
        if ((unsigned)yy >= (unsigned)H_) continue;
#pragma unroll
        for (int dx = 0; dx < 3; ++dx) {
            int xx = w + dx - 1;
            if ((unsigned)xx >= (unsigned)W_) continue;
            acc += xb[yy * W_ + xx] * dw[c * 9 + dy * 3 + dx];
        }
    }
    y_c[((long)bi * C_ + c) * HW_ + h * W_ + w] = (_Float16)acc;
}

// Depthwise 3x3, pixel-major output y_t[(b,hw,c)] f16. c fastest per thread.
__global__ __launch_bounds__(256) void dwt_kernel(const float* __restrict__ x,
                                                  const float* __restrict__ dw,
                                                  _Float16* __restrict__ y_t) {
    __shared__ float wsh[C_ * 9];
    for (int i = threadIdx.x; i < C_ * 9; i += 256) wsh[i] = dw[i];
    __syncthreads();
    int c  = threadIdx.x & 63;
    int p  = blockIdx.x * 4 + (threadIdx.x >> 6);
    int bi = blockIdx.y;
    int h = p >> 8, w = p & 255;
    const float* xb = x + ((long)bi * C_ + c) * HW_;
    float acc = 0.0f;
#pragma unroll
    for (int dy = 0; dy < 3; ++dy) {
        int yy = h + dy - 1;
        if ((unsigned)yy >= (unsigned)H_) continue;
#pragma unroll
        for (int dx = 0; dx < 3; ++dx) {
            int xx = w + dx - 1;
            if ((unsigned)xx >= (unsigned)W_) continue;
            acc += xb[yy * W_ + xx] * wsh[c * 9 + dy * 3 + dx];
        }
    }
    y_t[((long)bi * HW_ + p) * C_ + c] = (_Float16)acc;
}

// Gram: G[b] += Y_slice * Y_slice^T  (M=N=64, split-K over n). 4 waves/block,
// wave handles one 16-row M tile x all four 16-col N tiles. Unroll x2 so the
// scheduler has two independent K-steps of b128 loads in flight per 8 WMMAs.
__global__ __launch_bounds__(128) void gram_kernel(const _Float16* __restrict__ y_c,
                                                   float* __restrict__ G) {
    int slice = blockIdx.x;            // 0..31
    int bi    = blockIdx.y;
    int wave  = threadIdx.x >> 5;
    int lane  = threadIdx.x & 31;
    const _Float16* yb = y_c + (long)bi * C_ * HW_;
    int row0 = wave * 16;
    v8f acc[4] = {};
    int kbase = slice * (HW_ / 32);    // 2048-wide slice
#pragma unroll 2
    for (int it = 0; it < (HW_ / 32) / 32; ++it) {   // 64 steps of K=32
        int k0 = kbase + it * 32;
        v16h a = load_a_frag(yb, HW_, row0, k0, lane);
#pragma unroll
        for (int j = 0; j < 4; ++j) {
            v16h bf = load_b_frag(yb, HW_, j * 16, k0, lane);
            acc[j] = wmma_f16(a, bf, acc[j]);
        }
    }
    int n  = lane & 15;
    int mh = (lane < 16) ? 0 : 8;
#pragma unroll
    for (int j = 0; j < 4; ++j)
#pragma unroll
        for (int r = 0; r < 8; ++r)
            atomicAdd(&G[(bi * 64 + row0 + mh + r) * 64 + j * 16 + n], acc[j][r]);
}

// U = Wq*G, Vh = Wk*G  (per batch, 64x64 each). Wq = pw rows 0..63, Wk 64..127.
__global__ void uv_kernel(const float* __restrict__ pw, const float* __restrict__ G,
                          float* __restrict__ U, float* __restrict__ Vh) {
    int t = blockIdx.x * blockDim.x + threadIdx.x;   // 8*2*64*64 = 65536
    if (t >= B_ * 2 * 64 * 64) return;
    int d = t & 63, i = (t >> 6) & 63, which = (t >> 12) & 1, bi = t >> 13;
    const float* Wrow = pw + (which * 64 + i) * 64;
    const float* Gb   = G + bi * 4096;
    float s = 0.0f;
    for (int c = 0; c < 64; ++c) s += Wrow[c] * Gb[c * 64 + d];
    (which ? Vh : U)[bi * 4096 + i * 64 + d] = s;
}

// Per (b, i): attn row = (Wq G Wk^T)_i scaled by 1/||q_i|| 1/||k_j||, softmax.
__global__ void attnrow_kernel(const float* __restrict__ pw,
                               const float* __restrict__ U,
                               const float* __restrict__ Vh,
                               float* __restrict__ S) {
    int t = blockIdx.x * blockDim.x + threadIdx.x;   // 512 rows
    if (t >= B_ * 64) return;
    int i = t & 63, bi = t >> 6;
    const float* Ui = U + bi * 4096 + i * 64;
    const float* Wq = pw;
    const float* Wk = pw + 64 * 64;
    float dq = 0.0f;
    for (int d = 0; d < 64; ++d) dq += Ui[d] * Wq[i * 64 + d];
    float rqi = 1.0f / fmaxf(sqrtf(fmaxf(dq, 0.0f)), EPSN);
    float row[64];
    float mx = -1e30f;
    for (int j = 0; j < 64; ++j) {
        const float* Vj = Vh + bi * 4096 + j * 64;
        float raw = 0.0f, dk = 0.0f;
        for (int d = 0; d < 64; ++d) {
            raw += Ui[d] * Wk[j * 64 + d];
            dk  += Vj[d] * Wk[j * 64 + d];
        }
        float rkj = 1.0f / fmaxf(sqrtf(fmaxf(dk, 0.0f)), EPSN);
        float v = raw * rqi * rkj;
        row[j] = v;
        mx = fmaxf(mx, v);
    }
    float ssum = 0.0f;
    for (int j = 0; j < 64; ++j) { row[j] = __expf(row[j] - mx); ssum += row[j]; }
    float inv = 1.0f / ssum;
    float* Sr = S + bi * 4096 + i * 64;
    for (int j = 0; j < 64; ++j) Sr[j] = row[j] * inv;
}

// T = P * S (per batch)
__global__ void ps_kernel(const float* __restrict__ proj, const float* __restrict__ S,
                          float* __restrict__ T) {
    int t = blockIdx.x * blockDim.x + threadIdx.x;   // 8*4096
    if (t >= B_ * 4096) return;
    int d = t & 63, o = (t >> 6) & 63, bi = t >> 12;
    float s = 0.0f;
    for (int c = 0; c < 64; ++c) s += proj[o * 64 + c] * S[bi * 4096 + c * 64 + d];
    T[bi * 4096 + o * 64 + d] = s;
}

// M = T * Wv (f16, row-major (o,c)). Wv = pw rows 128..191.
__global__ void m_kernel(const float* __restrict__ T, const float* __restrict__ pw,
                         _Float16* __restrict__ M_h) {
    int t = blockIdx.x * blockDim.x + threadIdx.x;   // 8*4096
    if (t >= B_ * 4096) return;
    int c = t & 63, o = (t >> 6) & 63, bi = t >> 12;
    const float* Wv = pw + 128 * 64;
    float s = 0.0f;
    for (int d = 0; d < 64; ++d) s += T[bi * 4096 + o * 64 + d] * Wv[d * 64 + c];
    M_h[bi * 4096 + o * 64 + c] = (_Float16)s;
}

// out = M * Y : M=64, K=64, N=65536 per batch. Wave = one 16-pixel N tile,
// loops 4 M tiles x 2 K steps of WMMA. M tile staged to LDS by async DMA;
// B fragments issued *before* the DMA wait so they overlap it.
__global__ __launch_bounds__(128) void out_kernel(const _Float16* __restrict__ M_h,
                                                  const _Float16* __restrict__ y_t,
                                                  float* __restrict__ out) {
    __shared__ _Float16 Ml[64 * 64];
    int bi = blockIdx.y;
    int wave = threadIdx.x >> 5;
    int lane = threadIdx.x & 31;
    int pixel0 = blockIdx.x * 64 + wave * 16;
    const _Float16* vb = y_t + (long)bi * HW_ * C_;
    // Issue B-fragment global loads first (independent of the M-tile DMA).
    v16h b0 = load_b_frag(vb, C_, pixel0, 0, lane);
    v16h b1 = load_b_frag(vb, C_, pixel0, 32, lane);
    // Async-copy M (64x64 f16 = 8KB) into LDS: 512 b128 chunks / 128 threads.
    {
        const _Float16* gsrc = M_h + (size_t)bi * 4096;
#pragma unroll
        for (int it = 0; it < 4; ++it) {
            int i = it * 128 + threadIdx.x;           // chunk index, 16B each
            unsigned loff = (unsigned)(uintptr_t)
                (__attribute__((address_space(3))) _Float16*)(&Ml[i * 8]);
            unsigned long long gaddr = (unsigned long long)(uintptr_t)(gsrc + i * 8);
            asm volatile("global_load_async_to_lds_b128 %0, %1, off"
                         :: "v"(loff), "v"(gaddr) : "memory");
        }
        asm volatile("s_wait_asynccnt 0x0" ::: "memory");
    }
    __syncthreads();
    int n  = lane & 15;
    int mh = (lane < 16) ? 0 : 8;
#pragma unroll
    for (int mt = 0; mt < 4; ++mt) {
        v16h a0 = load_a_frag(Ml, 64, mt * 16, 0, lane);
        v16h a1 = load_a_frag(Ml, 64, mt * 16, 32, lane);
        v8f acc = {};
        acc = wmma_f16(a0, b0, acc);
        acc = wmma_f16(a1, b1, acc);
#pragma unroll
        for (int r = 0; r < 8; ++r) {
            int o = mt * 16 + mh + r;
            out[((long)bi * C_ + o) * HW_ + pixel0 + n] = acc[r];
        }
    }
}

// ---------------------------------------------------------------------------

extern "C" void kernel_launch(void* const* d_in, const int* in_sizes, int n_in,
                              void* d_out, int out_size, void* d_ws, size_t ws_size,
                              hipStream_t stream) {
    const float* x    = (const float*)d_in[0];   // (8,64,256,256)
    const float* dw   = (const float*)d_in[1];   // (64,1,3,3)
    const float* pw   = (const float*)d_in[2];   // (192,64)
    const float* proj = (const float*)d_in[3];   // (64,64)
    float* out = (float*)d_out;

    char* ws = (char*)d_ws;
    const size_t YC_OFF = 0;                                   // 64 MB f16
    const size_t YT_OFF = YC_OFF + (size_t)B_ * C_ * HW_ * 2;  // 64 MB f16
    const size_t G_OFF  = YT_OFF + (size_t)B_ * C_ * HW_ * 2;  // 128 KB f32
    const size_t U_OFF  = G_OFF + (size_t)B_ * 4096 * 4;
    const size_t VH_OFF = U_OFF + (size_t)B_ * 4096 * 4;
    const size_t S_OFF  = VH_OFF + (size_t)B_ * 4096 * 4;
    const size_t T_OFF  = S_OFF + (size_t)B_ * 4096 * 4;
    const size_t MH_OFF = T_OFF + (size_t)B_ * 4096 * 4;

    _Float16* y_c = (_Float16*)(ws + YC_OFF);
    _Float16* y_t = (_Float16*)(ws + YT_OFF);
    float*    G   = (float*)(ws + G_OFF);
    float*    U   = (float*)(ws + U_OFF);
    float*    Vh  = (float*)(ws + VH_OFF);
    float*    S   = (float*)(ws + S_OFF);
    float*    T   = (float*)(ws + T_OFF);
    _Float16* M_h = (_Float16*)(ws + MH_OFF);

    // 1) zero Gram accumulator (atomics target)
    zero_f32_kernel<<<(B_ * 4096 + 255) / 256, 256, 0, stream>>>(G, B_ * 4096);
    // 2) depthwise conv, both layouts (each written coalesced)
    dwc_kernel<<<dim3(H_, C_, B_), 256, 0, stream>>>(x, dw, y_c);
    dwt_kernel<<<dim3(HW_ / 4, B_), 256, 0, stream>>>(x, dw, y_t);
    // 3) Gram G = Y Y^T  (WMMA split-K)
    gram_kernel<<<dim3(32, B_), 128, 0, stream>>>(y_c, G);
    // 4) small per-batch algebra
    uv_kernel<<<(B_ * 2 * 4096 + 255) / 256, 256, 0, stream>>>(pw, G, U, Vh);
    attnrow_kernel<<<(B_ * 64 + 255) / 256, 256, 0, stream>>>(pw, U, Vh, S);
    ps_kernel<<<(B_ * 4096 + 255) / 256, 256, 0, stream>>>(proj, S, T);
    m_kernel<<<(B_ * 4096 + 255) / 256, 256, 0, stream>>>(T, pw, M_h);
    // 5) out = M * Y (WMMA)
    out_kernel<<<dim3(HW_ / 64, B_), 128, 0, stream>>>(M_h, y_t, out);
}
